// StockMambaL1_14748917695059
// MI455X (gfx1250) — compile-verified, matrix operand order checked by reading, not run on hardware
//
#include <hip/hip_runtime.h>
#include <hip/hip_bf16.h>
#include <math.h>

// ---------------------------------------------------------------------------
// CDNA5 (gfx1250) forward pass for the StockMamba reference.
// All matmuls run through V_WMMA_F32_16X16X32_F16 (fp32 -> f16 upcast, fp32 acc).
// GEMM: LDS-staged 128x64 block tile, 8 waves, 4 WMMA per wave per K-step.
// ---------------------------------------------------------------------------

typedef __attribute__((ext_vector_type(16))) _Float16 v16h;
typedef __attribute__((ext_vector_type(8)))  float    v8f;

#define NN    1024
#define BARS  64
#define FF    32
#define HH    256
#define HEADS 8
#define HD    32
#define ENC_L 4
#define FFN   1024
#define DD    512
#define DS_   16
#define DC_   4
#define ML_   8
#define DI_   1024
#define DR_   32
#define HID_  128

__device__ __forceinline__ float gelu_exact(float v) {
    return 0.5f * v * (1.0f + erff(v * 0.70710678118654752f));
}
__device__ __forceinline__ float silu_f(float v) {
    return v / (1.0f + expf(-v));
}

enum { ACT_NONE = 0, ACT_GELU = 1, ACT_SILU = 2, ACT_SOFTPLUS = 3 };

template <int ACT>
__device__ __forceinline__ float act_apply(float v) {
    if (ACT == ACT_GELU)     return gelu_exact(v);
    if (ACT == ACT_SILU)     return silu_f(v);
    if (ACT == ACT_SOFTPLUS) return (v > 20.0f) ? v : log1pf(expf(v));
    return v;
}

// ---------------------------------------------------------------------------
// LDS-tiled WMMA GEMM: C[M,N] = act(A[M,K] @ B[K,N] + bias + addsrc + pos)
// Block = 256 threads = 8 waves; block tile = 128(M) x 64(N); K-step = 32.
// Wave w owns rows [w*16, w*16+16) of the tile and all 64 cols (4 WMMAs/step,
// A fragment reused 4x). A staged row-major f16; B staged TRANSPOSED f16 so
// both fragment reads are contiguous 16B ds_load_b128s.
// grid = (N/64, M/128). Requires K%32==0, M%128==0, N%64==0 (all shapes here).
// ---------------------------------------------------------------------------
template <int ACT>
__global__ void gemm_wmma(const float* __restrict__ A, int lda,
                          const float* __restrict__ B, int ldb,
                          const float* __restrict__ bias,
                          const float* __restrict__ addsrc,
                          const float* __restrict__ pos, int posmask,
                          float* __restrict__ C, int ldc,
                          int M, int N, int K) {
    (void)M; (void)N;
    __shared__ _Float16 Asl[128 * 32];   // [row][k]   8 KB
    __shared__ _Float16 Bsl[64 * 32];    // [n][k]     4 KB (B transposed)

    const int tid  = threadIdx.x;
    const int lane = tid & 31;
    const int wv   = tid >> 5;
    const int tm0  = blockIdx.y << 7;    // * 128
    const int tn0  = blockIdx.x << 6;    // * 64

    v8f acc[4];
#pragma unroll
    for (int nt = 0; nt < 4; ++nt)
        acc[nt] = (v8f){0.f, 0.f, 0.f, 0.f, 0.f, 0.f, 0.f, 0.f};

    // staging coordinates (fixed per thread)
    const int arow = tid >> 1, akoff = (tid & 1) << 4;          // A: 16 elems/thread
    const int bk   = tid >> 3, bnoff = (tid & 7) << 3;          // B: 8 elems/thread

    for (int k0 = 0; k0 < K; k0 += 32) {
        // ---- stage A tile: 128 x 32, coalesced global reads ----
        {
            const float* Ap = A + (size_t)(tm0 + arow) * lda + k0 + akoff;
#pragma unroll
            for (int i = 0; i < 16; ++i)
                Asl[arow * 32 + akoff + i] = (_Float16)Ap[i];
        }
        // ---- stage B tile transposed: B[k0+bk][tn0+bnoff..+8] -> Bsl[n][k] ----
        {
            const float* Bp = B + (size_t)(k0 + bk) * ldb + tn0 + bnoff;
#pragma unroll
            for (int i = 0; i < 8; ++i)
                Bsl[(bnoff + i) * 32 + bk] = (_Float16)Bp[i];
        }
        if (k0 + 32 < K) {               // gfx1250 global_prefetch_b8 of next chunk
            __builtin_prefetch(A + (size_t)(tm0 + arow) * lda + k0 + 32 + akoff, 0, 1);
            __builtin_prefetch(B + (size_t)(k0 + 32 + bk) * ldb + tn0 + bnoff, 0, 1);
        }
        __syncthreads();

        // ---- compute: wave wv -> 16-row strip, 4 N-subtiles ----
        const int mA   = (wv << 4) + (lane & 15);
        const int ksub = (lane >> 4) << 3;   // 0 or 8
        v16h a;
#pragma unroll
        for (int i = 0; i < 8; ++i) {
            a[i]     = Asl[mA * 32 + ksub + i];
            a[i + 8] = Asl[mA * 32 + ksub + 16 + i];
        }
#pragma unroll
        for (int nt = 0; nt < 4; ++nt) {
            const int nB = (nt << 4) + (lane & 15);
            v16h b;
#pragma unroll
            for (int i = 0; i < 8; ++i) {
                b[i]     = Bsl[nB * 32 + ksub + i];
                b[i + 8] = Bsl[nB * 32 + ksub + 16 + i];
            }
            acc[nt] = __builtin_amdgcn_wmma_f32_16x16x32_f16(
                false, a, false, b, (short)0, acc[nt], false, false);
        }
        __syncthreads();
    }

    // ---- epilogue ----
    const int mC0 = tm0 + (wv << 4) + ((lane >> 4) << 3);
#pragma unroll
    for (int nt = 0; nt < 4; ++nt) {
        const int nC = tn0 + (nt << 4) + (lane & 15);
        const float bv = bias ? bias[nC] : 0.0f;
#pragma unroll
        for (int r = 0; r < 8; ++r) {
            const int m = mC0 + r;
            float v = acc[nt][r] + bv;
            if (addsrc) v += addsrc[(size_t)m * ldc + nC];
            if (pos)    v += pos[(size_t)(m & posmask) * ldc + nC];
            C[(size_t)m * ldc + nC] = act_apply<ACT>(v);
        }
    }
}

// ---------------------------------------------------------------------------
// Attention: one workgroup (256 thr = 8 waves) per (sample n, head h).
// q,k,v are 64x32 slices of the qkv buffer (row = n*64+bar, 768 cols).
// S = q k^T * scale + mask-bias (WMMA), softmax, O = P v (WMMA).
// ---------------------------------------------------------------------------
__global__ void attn_kernel(const float* __restrict__ qkv,
                            const float* __restrict__ mask,
                            float* __restrict__ o) {
    __shared__ _Float16 qs[BARS * HD];
    __shared__ _Float16 km[BARS * HD];
    __shared__ _Float16 vm[BARS * HD];
    __shared__ float    Ssc[BARS * BARS];
    __shared__ _Float16 Ph[BARS * BARS];
    __shared__ float    mrow[BARS];

    const int nidx = blockIdx.x >> 3;
    const int h    = blockIdx.x & 7;
    const float* base = qkv + (size_t)nidx * BARS * (3 * HH) + h * HD;

    for (int i = threadIdx.x; i < BARS * HD; i += 256) {
        const int bar = i >> 5, d = i & 31;
        const float* rp = base + (size_t)bar * (3 * HH) + d;
        qs[i] = (_Float16)rp[0];
        km[i] = (_Float16)rp[HH];
        vm[i] = (_Float16)rp[2 * HH];
    }
    if (threadIdx.x < BARS)
        mrow[threadIdx.x] = (1.0f - mask[nidx * BARS + threadIdx.x]) * -1.0e9f;
    __syncthreads();

    const int lane = threadIdx.x & 31;
    const int wv   = threadIdx.x >> 5;
    const int ksub = (lane >> 4) << 3;
    const float scale = 0.17677669529663687f;  // 1/sqrt(32)

    // ---- S = q @ k^T : 16 tiles of 16x16, K = 32 (one WMMA step) ----
    for (int iter = 0; iter < 2; ++iter) {
        const int t  = wv + 8 * iter;
        const int tm = (t >> 2) << 4, tn = (t & 3) << 4;
        const int mA = tm + (lane & 15), nB = tn + (lane & 15);
        v16h a, b;
        v8f acc = {0.f, 0.f, 0.f, 0.f, 0.f, 0.f, 0.f, 0.f};
#pragma unroll
        for (int i = 0; i < 8; ++i) {
            a[i]     = qs[mA * HD + ksub + i];
            a[i + 8] = qs[mA * HD + ksub + 16 + i];
            b[i]     = km[nB * HD + ksub + i];      // B[k][n] = k^T -> km[n][k]
            b[i + 8] = km[nB * HD + ksub + 16 + i];
        }
        acc = __builtin_amdgcn_wmma_f32_16x16x32_f16(
            false, a, false, b, (short)0, acc, false, false);
        const int mC0 = tm + ((lane >> 4) << 3), nC = tn + (lane & 15);
#pragma unroll
        for (int r = 0; r < 8; ++r)
            Ssc[(mC0 + r) * BARS + nC] = acc[r] * scale + mrow[nC];
    }
    __syncthreads();

    // ---- row softmax (64 rows, one thread each) ----
    if (threadIdx.x < BARS) {
        float* row = &Ssc[threadIdx.x * BARS];
        float mx = row[0];
        for (int j = 1; j < BARS; ++j) mx = fmaxf(mx, row[j]);
        float s = 0.f;
        for (int j = 0; j < BARS; ++j) { float e = expf(row[j] - mx); row[j] = e; s += e; }
        const float inv = 1.0f / s;
        for (int j = 0; j < BARS; ++j)
            Ph[threadIdx.x * BARS + j] = (_Float16)(row[j] * inv);
    }
    __syncthreads();

    // ---- O = P @ v : 8 tiles (4 x 2), K = 64 (two WMMA steps) ----
    {
        const int tm = (wv >> 1) << 4, tn = (wv & 1) << 4;
        const int mA = tm + (lane & 15), nB = tn + (lane & 15);
        v8f acc = {0.f, 0.f, 0.f, 0.f, 0.f, 0.f, 0.f, 0.f};
        for (int k0 = 0; k0 < BARS; k0 += 32) {
            v16h a, b;
#pragma unroll
            for (int i = 0; i < 8; ++i) {
                const int ka = k0 + ksub + i, kb2 = k0 + ksub + 16 + i;
                a[i]     = Ph[mA * BARS + ka];
                a[i + 8] = Ph[mA * BARS + kb2];
                b[i]     = vm[ka * HD + nB];
                b[i + 8] = vm[kb2 * HD + nB];
            }
            acc = __builtin_amdgcn_wmma_f32_16x16x32_f16(
                false, a, false, b, (short)0, acc, false, false);
        }
        const int mC0 = tm + ((lane >> 4) << 3), nC = tn + (lane & 15);
#pragma unroll
        for (int r = 0; r < 8; ++r)
            o[((size_t)nidx * BARS + mC0 + r) * HH + h * HD + nC] = acc[r];
    }
}

// ---------------------------------------------------------------------------
// Row LayerNorm (len = 256 or 512); optional post-LN residual add.
// One block (256 thr) per row.
// ---------------------------------------------------------------------------
__global__ void ln_kernel(const float* __restrict__ in,
                          const float* __restrict__ g, const float* __restrict__ b,
                          const float* __restrict__ res,
                          float* __restrict__ out, int len) {
    const int row = blockIdx.x, tid = threadIdx.x;
    const int per = len >> 8;
    const float* ip = in + (size_t)row * len;
    float vals[2]; float s = 0.f, s2 = 0.f;
    for (int i = 0; i < per; ++i) {
        float v = ip[tid + (i << 8)];
        vals[i] = v; s += v; s2 += v * v;
    }
    __shared__ float rs[256], rs2[256];
    rs[tid] = s; rs2[tid] = s2; __syncthreads();
    for (int st = 128; st > 0; st >>= 1) {
        if (tid < st) { rs[tid] += rs[tid + st]; rs2[tid] += rs2[tid + st]; }
        __syncthreads();
    }
    const float mean = rs[0] / (float)len;
    const float var  = rs2[0] / (float)len - mean * mean;
    const float rstd = rsqrtf(var + 1e-5f);
    float* op = out + (size_t)row * len;
    for (int i = 0; i < per; ++i) {
        const int c = tid + (i << 8);
        float v = (vals[i] - mean) * rstd * g[c] + b[c];
        if (res) v += res[(size_t)row * len + c];
        op[c] = v;
    }
}

// Masked mean over bars: frame_emb[n,d] = sum_bar x*m / max(sum m, 1)
__global__ void frame_pool_kernel(const float* __restrict__ x,
                                  const float* __restrict__ mask,
                                  float* __restrict__ out) {
    const int n = blockIdx.x, d = threadIdx.x;
    float s = 0.f, ms = 0.f;
    for (int bar = 0; bar < BARS; ++bar) {
        const float m = mask[n * BARS + bar];
        ms += m;
        s += x[((size_t)n * BARS + bar) * HH + d] * m;
    }
    out[(size_t)n * HH + d] = s / fmaxf(ms, 1.0f);
}

// Causal depthwise conv (width 4) + SiLU.  xi = xz[:, 0:1024] (ld 2048).
__global__ void conv_silu_kernel(const float* __restrict__ xz,
                                 const float* __restrict__ w,
                                 const float* __restrict__ b,
                                 float* __restrict__ xc) {
    const int i = blockIdx.x * 256 + threadIdx.x;    // 1024*1024
    const int t = i >> 10, c = i & 1023;
    float a = b[c];
#pragma unroll
    for (int k = 0; k < DC_; ++k) {
        const int tt = t - (DC_ - 1) + k;
        if (tt >= 0) a += xz[(size_t)tt * (2 * DI_) + c] * w[c * DC_ + k];
    }
    xc[i] = silu_f(a);
}

// Selective scan: one thread per channel c; h[16] in VGPRs; B/C via LDS.
__global__ void scan_kernel(const float* __restrict__ dt,
                            const float* __restrict__ xdbl,
                            const float* __restrict__ xc,
                            const float* __restrict__ Alog,
                            float* __restrict__ ys) {
    const int c = blockIdx.x * 256 + threadIdx.x;
    float h[DS_], A[DS_];
#pragma unroll
    for (int s = 0; s < DS_; ++s) { h[s] = 0.f; A[s] = -expf(Alog[(size_t)c * DS_ + s]); }
    __shared__ float Bs[DS_], Cs[DS_];
    for (int t = 0; t < NN; ++t) {
        if (threadIdx.x < 32) {
            const float v = xdbl[(size_t)t * 64 + DR_ + threadIdx.x];
            if (threadIdx.x < DS_) Bs[threadIdx.x] = v;
            else                   Cs[threadIdx.x - DS_] = v;
        }
        __syncthreads();
        const float d  = dt[(size_t)t * DI_ + c];
        const float xv = xc[(size_t)t * DI_ + c];
        float y = 0.f;
#pragma unroll
        for (int s = 0; s < DS_; ++s) {
            h[s] = h[s] * expf(d * A[s]) + d * Bs[s] * xv;
            y += h[s] * Cs[s];
        }
        ys[(size_t)t * DI_ + c] = y;
        __syncthreads();
    }
}

// y = (ys + D*xc) * silu(z),  z = xz[:, 1024:2048]
__global__ void gate_kernel(const float* __restrict__ ys,
                            const float* __restrict__ xc,
                            const float* __restrict__ Dv,
                            const float* __restrict__ xz,
                            float* __restrict__ y) {
    const int i = blockIdx.x * 256 + threadIdx.x;
    const int t = i >> 10, c = i & 1023;
    const float z = xz[(size_t)t * (2 * DI_) + DI_ + c];
    y[i] = (ys[i] + Dv[c] * xc[i]) * silu_f(z);
}

// logits[r] = x[r,:] . pool_w + pool_b   (1024 threads total)
__global__ void pool_logits_kernel(const float* __restrict__ x,
                                   const float* __restrict__ pw,
                                   const float* __restrict__ pb,
                                   float* __restrict__ logits) {
    const int r = blockIdx.x * 256 + threadIdx.x;
    float a = pb[0];
    const float* xp = x + (size_t)r * DD;
    for (int i = 0; i < DD; ++i) a += xp[i] * pw[i];
    logits[r] = a;
}

// softmax over 1024 rows, single block of 1024 threads
__global__ void softmax1024_kernel(const float* __restrict__ logits,
                                   float* __restrict__ w) {
    __shared__ float red[1024];
    const int tid = threadIdx.x;
    const float v = logits[tid];
    red[tid] = v; __syncthreads();
    for (int st = 512; st > 0; st >>= 1) {
        if (tid < st) red[tid] = fmaxf(red[tid], red[tid + st]);
        __syncthreads();
    }
    const float mx = red[0]; __syncthreads();
    const float e = expf(v - mx);
    red[tid] = e; __syncthreads();
    for (int st = 512; st > 0; st >>= 1) {
        if (tid < st) red[tid] += red[tid + st];
        __syncthreads();
    }
    w[tid] = e / red[0];
}

// hp[d] = sum_n w[n] * x[n,d]   (grid = 512 blocks)
__global__ void pool_weighted_kernel(const float* __restrict__ w,
                                     const float* __restrict__ x,
                                     float* __restrict__ hp) {
    const int d = blockIdx.x, tid = threadIdx.x;
    float s = 0.f;
    for (int n = tid; n < NN; n += 256) s += w[n] * x[(size_t)n * DD + d];
    __shared__ float red[256];
    red[tid] = s; __syncthreads();
    for (int st = 128; st > 0; st >>= 1) {
        if (tid < st) red[tid] += red[tid + st];
        __syncthreads();
    }
    if (tid == 0) hp[d] = red[0];
}

// Head MLP: vix = (gelu(ln(gelu(ln(hp@W1+b1))@W2+b2)))@W3+b3  (single block)
__global__ void head_kernel(const float* __restrict__ hp,
                            const float* __restrict__ w1, const float* __restrict__ b1,
                            const float* __restrict__ g1, const float* __restrict__ bb1,
                            const float* __restrict__ w2, const float* __restrict__ b2,
                            const float* __restrict__ g2, const float* __restrict__ bb2,
                            const float* __restrict__ w3, const float* __restrict__ b3,
                            float* __restrict__ out) {
    __shared__ float hs[DD], t1[HID_], t2[HID_], stat[2];
    const int tid = threadIdx.x;
    for (int i = tid; i < DD; i += 256) hs[i] = hp[i];
    __syncthreads();
    if (tid < HID_) {
        float a = b1[tid];
        for (int i = 0; i < DD; ++i) a += hs[i] * w1[i * HID_ + tid];
        t1[tid] = a;
    }
    __syncthreads();
    if (tid == 0) {
        float s = 0.f, s2 = 0.f;
        for (int i = 0; i < HID_; ++i) { s += t1[i]; s2 += t1[i] * t1[i]; }
        const float m = s / HID_;
        stat[0] = m; stat[1] = rsqrtf(s2 / HID_ - m * m + 1e-5f);
    }
    __syncthreads();
    if (tid < HID_)
        t1[tid] = gelu_exact((t1[tid] - stat[0]) * stat[1] * g1[tid] + bb1[tid]);
    __syncthreads();
    if (tid < HID_) {
        float a = b2[tid];
        for (int i = 0; i < HID_; ++i) a += t1[i] * w2[i * HID_ + tid];
        t2[tid] = a;
    }
    __syncthreads();
    if (tid == 0) {
        float s = 0.f, s2 = 0.f;
        for (int i = 0; i < HID_; ++i) { s += t2[i]; s2 += t2[i] * t2[i]; }
        const float m = s / HID_;
        stat[0] = m; stat[1] = rsqrtf(s2 / HID_ - m * m + 1e-5f);
    }
    __syncthreads();
    if (tid < HID_)
        t2[tid] = gelu_exact((t2[tid] - stat[0]) * stat[1] * g2[tid] + bb2[tid]);
    __syncthreads();
    if (tid == 0) {
        float a = b3[0];
        for (int i = 0; i < HID_; ++i) a += t2[i] * w3[i];
        out[0] = a;
    }
}

// ---------------------------------------------------------------------------
// Host orchestration
// ---------------------------------------------------------------------------
extern "C" void kernel_launch(void* const* d_in, const int* in_sizes, int n_in,
                              void* d_out, int out_size, void* d_ws, size_t ws_size,
                              hipStream_t stream) {
    (void)in_sizes; (void)n_in; (void)out_size; (void)ws_size;

    const float* frames   = (const float*)d_in[0];
    const float* fmask    = (const float*)d_in[1];
    const float* in_w     = (const float*)d_in[2];
    const float* in_b     = (const float*)d_in[3];
    const float* pos_emb  = (const float*)d_in[4];
    const float* qkv_w    = (const float*)d_in[5];
    const float* qkv_b    = (const float*)d_in[6];
    const float* ao_w     = (const float*)d_in[7];
    const float* ao_b     = (const float*)d_in[8];
    const float* ln1_g    = (const float*)d_in[9];
    const float* ln1_b    = (const float*)d_in[10];
    const float* f1_w     = (const float*)d_in[11];
    const float* f1_b     = (const float*)d_in[12];
    const float* f2_w     = (const float*)d_in[13];
    const float* f2_b     = (const float*)d_in[14];
    const float* ln2_g    = (const float*)d_in[15];
    const float* ln2_b    = (const float*)d_in[16];
    const float* proj_w   = (const float*)d_in[17];
    const float* proj_b   = (const float*)d_in[18];
    const float* m_in_w   = (const float*)d_in[19];
    const float* m_conv_w = (const float*)d_in[20];
    const float* m_conv_b = (const float*)d_in[21];
    const float* m_xp_w   = (const float*)d_in[22];
    const float* m_dt_w   = (const float*)d_in[23];
    const float* m_dt_b   = (const float*)d_in[24];
    const float* m_Alog   = (const float*)d_in[25];
    const float* m_D      = (const float*)d_in[26];
    const float* m_out_w  = (const float*)d_in[27];
    const float* m_ln_g   = (const float*)d_in[28];
    const float* m_ln_b   = (const float*)d_in[29];
    const float* pool_w   = (const float*)d_in[30];
    const float* pool_b   = (const float*)d_in[31];
    const float* h1_w     = (const float*)d_in[32];
    const float* h1_b     = (const float*)d_in[33];
    const float* hln1_g   = (const float*)d_in[34];
    const float* hln1_b   = (const float*)d_in[35];
    const float* h2_w     = (const float*)d_in[36];
    const float* h2_b     = (const float*)d_in[37];
    const float* hln2_g   = (const float*)d_in[38];
    const float* hln2_b   = (const float*)d_in[39];
    const float* h3_w     = (const float*)d_in[40];
    const float* h3_b     = (const float*)d_in[41];

    const int R = NN * BARS;   // 65536 token rows

    // ---- workspace carving (floats) ----
    float* ws  = (float*)d_ws;
    float* xb  = ws;                              // 65536 x 256
    float* big = xb + (size_t)R * HH;             // 65536 x 1024 (qkv / ffn / mamba)
    float* ob  = big + (size_t)R * FFN;           // 65536 x 256
    float* tb  = ob + (size_t)R * HH;             // 65536 x 256
    // mamba sub-buffers inside `big` (encoder use of `big` is done by then)
    float* x2   = big;                            // 1024 x 512
    float* xz   = x2   + (size_t)NN * DD;         // 1024 x 2048
    float* xc   = xz   + (size_t)NN * 2 * DI_;    // 1024 x 1024
    float* xdbl = xc   + (size_t)NN * DI_;        // 1024 x 64
    float* dtb  = xdbl + (size_t)NN * 64;         // 1024 x 1024
    float* ysb  = dtb  + (size_t)NN * DI_;        // 1024 x 1024
    float* yb   = ysb  + (size_t)NN * DI_;        // 1024 x 1024
    float* t2b  = yb   + (size_t)NN * DI_;        // 1024 x 512
    float* logits = t2b + (size_t)NN * DD;        // 1024
    float* wgt    = logits + NN;                  // 1024
    float* hp     = wgt + NN;                     // 512

    const dim3 blk(256);

    // ---- input projection: x = frames @ in_w + in_b + pos_emb ----
    gemm_wmma<ACT_NONE><<<dim3(HH / 64, R / 128), blk, 0, stream>>>(
        frames, FF, in_w, HH, in_b, nullptr, pos_emb, BARS - 1, xb, HH, R, HH, FF);

    // ---- encoder ----
    for (int l = 0; l < ENC_L; ++l) {
        gemm_wmma<ACT_NONE><<<dim3(3 * HH / 64, R / 128), blk, 0, stream>>>(
            xb, HH, qkv_w + (size_t)l * HH * 3 * HH, 3 * HH,
            qkv_b + (size_t)l * 3 * HH, nullptr, nullptr, 0,
            big, 3 * HH, R, 3 * HH, HH);

        attn_kernel<<<NN * HEADS, 256, 0, stream>>>(big, fmask, ob);

        gemm_wmma<ACT_NONE><<<dim3(HH / 64, R / 128), blk, 0, stream>>>(
            ob, HH, ao_w + (size_t)l * HH * HH, HH,
            ao_b + (size_t)l * HH, xb, nullptr, 0, tb, HH, R, HH, HH);

        ln_kernel<<<R, 256, 0, stream>>>(
            tb, ln1_g + (size_t)l * HH, ln1_b + (size_t)l * HH, nullptr, xb, HH);

        gemm_wmma<ACT_GELU><<<dim3(FFN / 64, R / 128), blk, 0, stream>>>(
            xb, HH, f1_w + (size_t)l * HH * FFN, FFN,
            f1_b + (size_t)l * FFN, nullptr, nullptr, 0, big, FFN, R, FFN, HH);

        gemm_wmma<ACT_NONE><<<dim3(HH / 64, R / 128), blk, 0, stream>>>(
            big, FFN, f2_w + (size_t)l * FFN * HH, HH,
            f2_b + (size_t)l * HH, xb, nullptr, 0, tb, HH, R, HH, FFN);

        ln_kernel<<<R, 256, 0, stream>>>(
            tb, ln2_g + (size_t)l * HH, ln2_b + (size_t)l * HH, nullptr, xb, HH);
    }

    // ---- masked frame pooling + projection to D ----
    frame_pool_kernel<<<NN, HH, 0, stream>>>(xb, fmask, ob);
    gemm_wmma<ACT_NONE><<<dim3(DD / 64, NN / 128), blk, 0, stream>>>(
        ob, HH, proj_w, DD, proj_b, nullptr, nullptr, 0, x2, DD, NN, DD, HH);

    // ---- mamba stack ----
    for (int l = 0; l < ML_; ++l) {
        gemm_wmma<ACT_NONE><<<dim3(2 * DI_ / 64, NN / 128), blk, 0, stream>>>(
            x2, DD, m_in_w + (size_t)l * DD * 2 * DI_, 2 * DI_,
            nullptr, nullptr, nullptr, 0, xz, 2 * DI_, NN, 2 * DI_, DD);

        conv_silu_kernel<<<NN * DI_ / 256, 256, 0, stream>>>(
            xz, m_conv_w + (size_t)l * DI_ * DC_, m_conv_b + (size_t)l * DI_, xc);

        gemm_wmma<ACT_NONE><<<dim3(64 / 64, NN / 128), blk, 0, stream>>>(
            xc, DI_, m_xp_w + (size_t)l * DI_ * 64, 64,
            nullptr, nullptr, nullptr, 0, xdbl, 64, NN, 64, DI_);

        gemm_wmma<ACT_SOFTPLUS><<<dim3(DI_ / 64, NN / 128), blk, 0, stream>>>(
            xdbl, 64, m_dt_w + (size_t)l * DR_ * DI_, DI_,
            m_dt_b + (size_t)l * DI_, nullptr, nullptr, 0, dtb, DI_, NN, DI_, DR_);

        scan_kernel<<<DI_ / 256, 256, 0, stream>>>(
            dtb, xdbl, xc, m_Alog + (size_t)l * DI_ * DS_, ysb);

        gate_kernel<<<NN * DI_ / 256, 256, 0, stream>>>(
            ysb, xc, m_D + (size_t)l * DI_, xz, yb);

        gemm_wmma<ACT_NONE><<<dim3(DD / 64, NN / 128), blk, 0, stream>>>(
            yb, DI_, m_out_w + (size_t)l * DI_ * DD, DD,
            nullptr, nullptr, nullptr, 0, t2b, DD, NN, DD, DI_);

        ln_kernel<<<NN, 256, 0, stream>>>(
            t2b, m_ln_g + (size_t)l * DD, m_ln_b + (size_t)l * DD, x2, x2, DD);
    }

    // ---- attention pooling over N + head MLP ----
    pool_logits_kernel<<<NN / 256, 256, 0, stream>>>(x2, pool_w, pool_b, logits);
    softmax1024_kernel<<<1, 1024, 0, stream>>>(logits, wgt);
    pool_weighted_kernel<<<DD, 256, 0, stream>>>(wgt, x2, hp);

    head_kernel<<<1, 256, 0, stream>>>(
        hp, h1_w, h1_b, hln1_g, hln1_b, h2_w, h2_b, hln2_g, hln2_b,
        h3_w, h3_b, (float*)d_out);

    // second output: final x (1, 1024, 512) flattened after vix
    hipMemcpyAsync((float*)d_out + 1, x2, (size_t)NN * DD * sizeof(float),
                   hipMemcpyDeviceToDevice, stream);
}
